// TFSA_1537598292582
// MI455X (gfx1250) — matrix operand order repaired and checked by reading, not compiled, not run on hardware
//
#include <hip/hip_runtime.h>
#include <hip/hip_bf16.h>
#include <stdint.h>

// ---------------------------------------------------------------------------
// TFSA for MI455X / gfx1250. bf16 WMMA (f32 accumulate) everywhere, bf16
// intermediates in per-use-transposed layouts (contiguous fragment loads),
// TDM (tensor_load_to_lds) staging of the K-strip in frequency attention.
// ---------------------------------------------------------------------------

typedef __attribute__((ext_vector_type(16))) __bf16 bf16x16;
typedef __attribute__((ext_vector_type(8)))  __bf16 bf16x8;
typedef __attribute__((ext_vector_type(8)))  float  floatx8;
typedef __attribute__((ext_vector_type(4)))  unsigned int uint32x4;
typedef __attribute__((ext_vector_type(8)))  int    int32x8;
typedef __attribute__((ext_vector_type(4)))  int    int32x4;

constexpr int BB = 4, CC = 64, FF = 257, TT = 400, DD = 16;
constexpr int FT   = FF * TT;    // 102800
constexpr int FP2  = 288;        // freq dim padded to 18*16
constexpr int VROW = 288;        // v inner stride (f), padded
constexpr int TTP  = 416;        // f_out inner stride (t), padded (13*32)
constexpr int NP3  = 416;        // time score strip

__device__ __forceinline__ floatx8 wmma_bf16(bf16x16 a, bf16x16 b, floatx8 c) {
    return __builtin_amdgcn_wmma_f32_16x16x32_bf16(false, a, false, b, (short)0, c,
                                                   false, false);
}

__device__ __forceinline__ float bn_prelu1(float x, float g, float be, float m,
                                           float v, float al) {
    float xh = (x - m) * rsqrtf(v + 1e-5f) * g + be;
    return xh > 0.0f ? xh : al * xh;
}

// 1-D contiguous bf16 strip: global -> LDS via Tensor Data Mover.
// D# packed per CDNA5 ISA 8.3/8.4: count=1, type=2(image), data_size=1(2B),
// tensor_dim0 = tile_dim0 = nelem, tensor_dim1 = tile_dim1 = 1.
// 6-arg builtin form: (g0, g1, g2, g3, g_extra, cpol).
__device__ __forceinline__ void tdm_load_1d(const void* gsrc, void* lds_dst,
                                            uint32_t nelem) {
    uint64_t ga = (uint64_t)(uintptr_t)gsrc;
    uint32_t la = (uint32_t)(uintptr_t)lds_dst;   // low 32 bits = LDS offset
    uint32x4 g0;
    g0[0] = 1u;                                               // count=1
    g0[1] = la;                                               // lds_addr
    g0[2] = (uint32_t)ga;                                     // global_addr[31:0]
    g0[3] = (uint32_t)((ga >> 32) & 0x01FFFFFFu) | (2u << 30);// ga[56:32] | type=2
    int32x8 g1;
    g1[0] = 1 << 16;                                          // data_size=1 (2B)
    g1[1] = (int)((nelem & 0xFFFFu) << 16);                   // tensor_dim0[15:0]
    g1[2] = (int)(((nelem >> 16) & 0xFFFFu) | (1u << 16));    // dim0 hi | tensor_dim1=1
    g1[3] = (int)((nelem & 0xFFFFu) << 16);                   // tile_dim0
    g1[4] = 1;                                                // tile_dim1=1
    g1[5] = (int)nelem;                                       // tensor_dim0_stride lo
    g1[6] = 0;
    g1[7] = 0;
    int32x4 z4 = {0, 0, 0, 0};
    int32x8 z8 = {0, 0, 0, 0, 0, 0, 0, 0};
    __builtin_amdgcn_tensor_load_to_lds(g0, g1, z4, z4, z8, 0);
}

// ---------------------------------------------------------------------------
// Kernel 1: fused QKV(48x64)+TQK(32x64) GEMM. One block = one 16-pos tile,
// all 5 M-tiles (input B-fragment loaded once, reused 5x). BN/PReLU epilogue,
// bf16 stores into per-use-transposed layouts.
// ---------------------------------------------------------------------------
__global__ __launch_bounds__(32) void k_qkv(
    const float* __restrict__ inp, const float* __restrict__ Wf,
    const float* __restrict__ Wt,
    const float* __restrict__ gf, const float* __restrict__ bf,
    const float* __restrict__ mf, const float* __restrict__ vf,
    const float* __restrict__ af,
    const float* __restrict__ gt, const float* __restrict__ bt,
    const float* __restrict__ mt, const float* __restrict__ vt,
    const float* __restrict__ at,
    __bf16* __restrict__ qf, __bf16* __restrict__ kf, __bf16* __restrict__ vv,
    __bf16* __restrict__ qt, __bf16* __restrict__ kt)
{
    const int lane = threadIdx.x, lid = lane & 15, half = lane >> 4;
    const int pos0 = blockIdx.x * 16;
    const int b    = blockIdx.z;
    const float* xcol = inp + (size_t)b * CC * FT + pos0 + lid;

    floatx8 acc[5] = {};
    for (int kc = 0; kc < CC; kc += 32) {
        bf16x16 bfrag;
#pragma unroll
        for (int j = 0; j < 16; ++j)
            bfrag[j] = (__bf16)xcol[(size_t)(kc + (j & 8) * 2 + half * 8 + (j & 7)) * FT];
#pragma unroll
        for (int m = 0; m < 5; ++m) {
            const int o = m * 16 + lid;
            const float* Wrow = (o < 48) ? (Wf + o * CC) : (Wt + (o - 48) * CC);
            bf16x16 afrag;
#pragma unroll
            for (int j = 0; j < 8; ++j) {
                afrag[j]     = (__bf16)Wrow[kc + half * 8 + j];
                afrag[8 + j] = (__bf16)Wrow[kc + 16 + half * 8 + j];
            }
            acc[m] = wmma_bf16(afrag, bfrag, acc[m]);
        }
    }

    const int pos = pos0 + lid;
    const int f = pos / TT, t = pos % TT;
#pragma unroll
    for (int m = 0; m < 5; ++m) {
#pragma unroll
        for (int r = 0; r < 8; ++r) {
            int oo = m * 16 + r + half * 8;
            float x = acc[m][r];
            if (oo < 48) {
                float y = bn_prelu1(x, gf[oo], bf[oo], mf[oo], vf[oo], af[oo]);
                __bf16 yb = (__bf16)y;
                int s = oo % 3, dc = oo / 3;
                if (s == 0)       qf[(((size_t)b * TT + t) * FF + f) * DD + dc] = yb;
                else if (s == 1)  kf[(((size_t)b * TT + t) * FF + f) * DD + dc] = yb;
                else              vv[(((size_t)b * TT + t) * DD + dc) * VROW + f] = yb;
            } else {
                int o2 = oo - 48;
                float y = bn_prelu1(x, gt[o2], bt[o2], mt[o2], vt[o2], at[o2]);
                __bf16 yb = (__bf16)y;
                int s = o2 % 2, dc = o2 / 2;
                size_t idx = (((size_t)b * FF + f) * TT + t) * DD + dc;
                if (s == 0) qt[idx] = yb; else kt[idx] = yb;
            }
        }
    }
}

// ---------------------------------------------------------------------------
// Kernel 2: frequency attention. K-strip staged to LDS via TDM.
// ---------------------------------------------------------------------------
__global__ __launch_bounds__(32) void k_fattn(
    const __bf16* __restrict__ qf, const __bf16* __restrict__ kf,
    const __bf16* __restrict__ v, __bf16* __restrict__ fout)
{
    __shared__ float S[16][FP2];
    __shared__ __align__(16) __bf16 Ksh[FF * DD];
    const int lane = threadIdx.x, lid = lane & 15, half = lane >> 4;
    const int ftile = blockIdx.x;   // 0..16
    const int t     = blockIdx.y;   // 0..399
    const int b     = blockIdx.z;

    // DMA the whole K slice for (b,t): contiguous 257*16 bf16.
    tdm_load_1d(kf + ((size_t)b * TT + t) * FF * DD, Ksh, FF * DD);

    // A: Q rows f (contiguous bf16), K = c 0..15 real, 16..31 zero-pad.
    int f  = ftile * 16 + lid;
    int fc = f > FF - 1 ? FF - 1 : f;
    bf16x16 afrag;
    {
        bf16x8 qv = *(const bf16x8*)(qf + (((size_t)b * TT + t) * FF + fc) * DD + half * 8);
#pragma unroll
        for (int j = 0; j < 8; ++j) { afrag[j] = qv[j]; afrag[8 + j] = (__bf16)0.0f; }
    }

    __builtin_amdgcn_s_wait_tensorcnt(0);
    __asm__ volatile("" ::: "memory");

    for (int yt = 0; yt < FP2 / 16; ++yt) {
        bf16x16 bfrag;
        int y  = yt * 16 + lid;
        int yc = y > FF - 1 ? FF - 1 : y;
        {
            bf16x8 lo = *(const bf16x8*)(&Ksh[yc * DD + half * 8]);
#pragma unroll
            for (int j = 0; j < 8; ++j) { bfrag[j] = lo[j]; bfrag[8 + j] = (__bf16)0.0f; }
        }
        floatx8 c = {};
        c = wmma_bf16(afrag, bfrag, c);
#pragma unroll
        for (int r = 0; r < 8; ++r) {
            int yg = yt * 16 + lid;
            float sv = c[r] * 0.25f;
            if (yg > FF - 1) sv = -1e30f;
            S[r + half * 8][yg] = sv;
        }
    }
    __syncthreads();

    if (half == 0) {
        int row = lid;
        float m = -1e30f;
        for (int y = 0; y < 272; ++y) m = fmaxf(m, S[row][y]);
        float s = 0.0f;
        for (int y = 0; y < 272; ++y) { float e = __expf(S[row][y] - m); S[row][y] = e; s += e; }
        float inv = 1.0f / s;
        for (int y = 0; y < 272; ++y) S[row][y] *= inv;
        for (int y = 272; y < FP2; ++y) S[row][y] = 0.0f;
    }
    __syncthreads();

    // O[f,c] = sum_y P[f,y] * V[c,y]; V rows contiguous over f (padded to 288).
    floatx8 acc = {};
    for (int yc0 = 0; yc0 < FP2; yc0 += 32) {
        bf16x16 pa, vb;
#pragma unroll
        for (int j = 0; j < 8; ++j) {
            pa[j]     = (__bf16)S[lid][yc0 + half * 8 + j];
            pa[8 + j] = (__bf16)S[lid][yc0 + 16 + half * 8 + j];
        }
        const __bf16* vr = v + (((size_t)b * TT + t) * DD + lid) * VROW + yc0 + half * 8;
        bf16x8 lo = *(const bf16x8*)vr;
        bf16x8 hi = *(const bf16x8*)(vr + 16);
#pragma unroll
        for (int j = 0; j < 8; ++j) { vb[j] = lo[j]; vb[8 + j] = hi[j]; }
        acc = wmma_bf16(pa, vb, acc);
    }
#pragma unroll
    for (int r = 0; r < 8; ++r) {
        int fg = ftile * 16 + r + half * 8;
        if (fg < FF)
            fout[(((size_t)b * FF + fg) * DD + lid) * TTP + t] = (__bf16)acc[r];
    }
}

// ---------------------------------------------------------------------------
// Kernel 3: causal time attention over f_out (lower-triangular tiles only).
// ---------------------------------------------------------------------------
__global__ __launch_bounds__(32) void k_tattn(
    const __bf16* __restrict__ qt, const __bf16* __restrict__ kt,
    const __bf16* __restrict__ fout, __bf16* __restrict__ tout)
{
    __shared__ float S[16][NP3];
    const int lane = threadIdx.x, lid = lane & 15, half = lane >> 4;
    const int ttile = blockIdx.x;   // 0..24
    const int f     = blockIdx.y;   // 0..256
    const int b     = blockIdx.z;
    const int t0  = ttile * 16;
    const int nyt = ttile + 1;
    const int L   = nyt * 16;
    const int Lp  = (L + 31) & ~31;
    const size_t rowbase = ((size_t)b * FF + f);

    bf16x16 afrag;
    {
        bf16x8 qv = *(const bf16x8*)(qt + (rowbase * TT + t0 + lid) * DD + half * 8);
#pragma unroll
        for (int j = 0; j < 8; ++j) { afrag[j] = qv[j]; afrag[8 + j] = (__bf16)0.0f; }
    }

    for (int yt = 0; yt < nyt; ++yt) {
        bf16x16 bfrag;
        {
            const __bf16* kr = kt + (rowbase * TT + yt * 16 + lid) * DD + half * 8;
            bf16x8 lo = *(const bf16x8*)kr;
#pragma unroll
            for (int j = 0; j < 8; ++j) { bfrag[j] = lo[j]; bfrag[8 + j] = (__bf16)0.0f; }
        }
        floatx8 c = {};
        c = wmma_bf16(afrag, bfrag, c);
#pragma unroll
        for (int r = 0; r < 8; ++r) {
            int tg = t0 + r + half * 8;
            int yg = yt * 16 + lid;
            float sv = c[r] * 0.25f;
            if (yg > tg) sv = -1e30f;
            S[r + half * 8][yg] = sv;
        }
    }
    __syncthreads();

    if (half == 0) {
        int row = lid;
        float m = -1e30f;
        for (int y = 0; y < L; ++y) m = fmaxf(m, S[row][y]);
        float s = 0.0f;
        for (int y = 0; y < L; ++y) { float e = __expf(S[row][y] - m); S[row][y] = e; s += e; }
        float inv = 1.0f / s;
        for (int y = 0; y < L; ++y) S[row][y] *= inv;
        for (int y = L; y < Lp; ++y) S[row][y] = 0.0f;
    }
    __syncthreads();

    floatx8 acc = {};
    for (int yc0 = 0; yc0 < Lp; yc0 += 32) {
        bf16x16 pa, vb;
#pragma unroll
        for (int j = 0; j < 8; ++j) {
            pa[j]     = (__bf16)S[lid][yc0 + half * 8 + j];
            pa[8 + j] = (__bf16)S[lid][yc0 + 16 + half * 8 + j];
        }
        const __bf16* vr = fout + (rowbase * DD + lid) * TTP + yc0 + half * 8;
        bf16x8 lo = *(const bf16x8*)vr;
        bf16x8 hi = *(const bf16x8*)(vr + 16);
#pragma unroll
        for (int j = 0; j < 8; ++j) { vb[j] = lo[j]; vb[8 + j] = hi[j]; }
        acc = wmma_bf16(pa, vb, acc);
    }
#pragma unroll
    for (int r = 0; r < 8; ++r) {
        int tg = t0 + r + half * 8;
        tout[((size_t)b * FT + f * TT + tg) * DD + lid] = (__bf16)acc[r];
    }
}

// ---------------------------------------------------------------------------
// Kernel 4: output projection Wp (64x16) + BN/PReLU + residual.
// One block = one 16-pos tile, all 4 M-tiles share one B-fragment.
// ---------------------------------------------------------------------------
__global__ __launch_bounds__(32) void k_proj(
    const __bf16* __restrict__ tout, const float* __restrict__ Wp,
    const float* __restrict__ gp, const float* __restrict__ bp,
    const float* __restrict__ mp, const float* __restrict__ vp,
    const float* __restrict__ ap,
    const float* __restrict__ inp, float* __restrict__ out)
{
    const int lane = threadIdx.x, lid = lane & 15, half = lane >> 4;
    const int pos = blockIdx.x * 16 + lid;
    const int b   = blockIdx.z;

    bf16x16 bfrag;
    {
        const __bf16* tr = tout + ((size_t)b * FT + pos) * DD + half * 8;
        bf16x8 lo = *(const bf16x8*)tr;
#pragma unroll
        for (int j = 0; j < 8; ++j) { bfrag[j] = lo[j]; bfrag[8 + j] = (__bf16)0.0f; }
    }

#pragma unroll
    for (int m = 0; m < 4; ++m) {
        const int o = m * 16 + lid;
        bf16x16 afrag;
#pragma unroll
        for (int j = 0; j < 8; ++j) {
            afrag[j]     = (__bf16)Wp[o * DD + half * 8 + j];
            afrag[8 + j] = (__bf16)0.0f;
        }
        floatx8 c = {};
        c = wmma_bf16(afrag, bfrag, c);
#pragma unroll
        for (int r = 0; r < 8; ++r) {
            int oo = m * 16 + r + half * 8;
            float y = bn_prelu1(c[r], gp[oo], bp[oo], mp[oo], vp[oo], ap[oo]);
            size_t idx = ((size_t)b * CC + oo) * FT + pos;
            out[idx] = y + inp[idx];
        }
    }
}

// ---------------------------------------------------------------------------
extern "C" void kernel_launch(void* const* d_in, const int* in_sizes, int n_in,
                              void* d_out, int out_size, void* d_ws, size_t ws_size,
                              hipStream_t stream)
{
    const float* inp = (const float*)d_in[0];
    const float* Wf  = (const float*)d_in[1];
    const float* gf  = (const float*)d_in[2];
    const float* bf  = (const float*)d_in[3];
    const float* mf  = (const float*)d_in[4];
    const float* vf  = (const float*)d_in[5];
    const float* af  = (const float*)d_in[6];
    const float* Wt  = (const float*)d_in[7];
    const float* gt  = (const float*)d_in[8];
    const float* bt  = (const float*)d_in[9];
    const float* mt  = (const float*)d_in[10];
    const float* vt  = (const float*)d_in[11];
    const float* at  = (const float*)d_in[12];
    const float* Wp  = (const float*)d_in[13];
    const float* gp  = (const float*)d_in[14];
    const float* bp  = (const float*)d_in[15];
    const float* mp  = (const float*)d_in[16];
    const float* vp  = (const float*)d_in[17];
    const float* ap  = (const float*)d_in[18];
    float* out = (float*)d_out;

    __bf16* ws = (__bf16*)d_ws;
    const size_t n_qf   = (size_t)BB * TT * FF * DD;   // [b][t][f][c]
    const size_t n_kf   = n_qf;                        // [b][t][f][c]
    const size_t n_v    = (size_t)BB * TT * DD * VROW; // [b][t][c][fpad]
    const size_t n_qt   = (size_t)BB * FF * TT * DD;   // [b][f][t][c]
    const size_t n_kt   = n_qt;                        // [b][f][t][c]
    const size_t n_fout = (size_t)BB * FF * DD * TTP;  // [b][f][c][tpad]

    __bf16* qf   = ws;
    __bf16* kf   = qf + n_qf;
    __bf16* v    = kf + n_kf;
    __bf16* qt   = v + n_v;
    __bf16* kt   = qt + n_qt;
    __bf16* fout = kt + n_kt;
    __bf16* tout = fout + n_fout;                      // [b][pos][c]

    dim3 g1(FT / 16, 1, BB);
    k_qkv<<<g1, 32, 0, stream>>>(inp, Wf, Wt, gf, bf, mf, vf, af,
                                 gt, bt, mt, vt, at, qf, kf, v, qt, kt);

    dim3 g2((FF + 15) / 16, TT, BB);
    k_fattn<<<g2, 32, 0, stream>>>(qf, kf, v, fout);

    dim3 g3(TT / 16, FF, BB);
    k_tattn<<<g3, 32, 0, stream>>>(qt, kt, fout, tout);

    dim3 g4(FT / 16, 1, BB);
    k_proj<<<g4, 32, 0, stream>>>(tout, Wp, gp, bp, mp, vp, ap, inp, out);
}
